// RepCodec_53618371723474
// MI455X (gfx1250) — matrix operand order repaired and checked by reading, not compiled, not run on hardware
//
#include <hip/hip_runtime.h>
#include <math.h>

// ---------------- problem constants ----------------
static constexpr int BB  = 8;
static constexpr int TT  = 2048;
static constexpr int HH  = 1024;
static constexpr int DD  = 384;
static constexpr int II  = 2048;
static constexpr int LL  = 12;
static constexpr int CBN = 8192;
static constexpr int CDM = 8;
static constexpr int MT  = BB * TT;      // 16384 rows
static constexpr int TP  = TT + 6;       // padded time

// ---------------- types ----------------
typedef __bf16 bf16;
typedef bf16  v16bf __attribute__((ext_vector_type(16)));
typedef bf16  v8bf  __attribute__((ext_vector_type(8)));
typedef float v8f   __attribute__((ext_vector_type(8)));

__device__ __forceinline__ v16bf frag_cat(v8bf lo, v8bf hi) {
  return __builtin_shufflevector(lo, hi, 0,1,2,3,4,5,6,7,8,9,10,11,12,13,14,15);
}

// A fragment (16x32 bf16, M x K), A row-major with leading dim lda.
// Lanes 0-15: row M=lane, K = {k0..k0+7} U {k0+16..k0+23}
// Lanes 16-31: row M=lane-16, K = {k0+8..k0+15} U {k0+24..k0+31}
__device__ __forceinline__ v16bf load_a_frag(const bf16* __restrict__ A, int lda,
                                             int row, int k0, int lane) {
  const bf16* p = A + (size_t)row * lda + k0 + ((lane & 16) ? 8 : 0);
  return frag_cat(*(const v8bf*)p, *(const v8bf*)(p + 16));
}

// B fragment (32x16 bf16, K x N) where W is stored (N, K) row-major (i.e. we
// compute A @ W^T).  Lanes 0-15: col N, K = {k0..k0+7} U {k0+16..k0+23}; lanes 16-31: K = {k0+8..k0+15} U {k0+24..k0+31}.
__device__ __forceinline__ v16bf load_b_frag(const bf16* __restrict__ W, int ldb,
                                             int col, int k0, int lane) {
  const bf16* p = W + (size_t)col * ldb + k0 + ((lane & 16) ? 8 : 0);
  __builtin_prefetch(p + 256, 0, 1);      // stream weight panel ahead (global_prefetch_b8)
  return frag_cat(*(const v8bf*)p, *(const v8bf*)(p + 16));
}

__device__ __forceinline__ float gelu_exact(float v) {
  return 0.5f * v * (1.0f + erff(v * 0.70710678118654752f));
}

// ---------------- packing / casting kernels ----------------
__global__ void build_xpad(const float* __restrict__ x, bf16* __restrict__ xp) {
  int i = blockIdx.x * blockDim.x + threadIdx.x;
  const int total = BB * TP * HH;
  if (i >= total) return;
  int b   = i / (TP * HH);
  int rem = i - b * (TP * HH);
  int tp  = rem >> 10;          // /HH
  int c   = rem & (HH - 1);
  float v = 0.0f;
  if (tp >= 3 && tp < TT + 3)
    v = x[((size_t)b * TT + (tp - 3)) * HH + c];
  xp[i] = (bf16)v;
}

// ew: (D, H, 7) -> wek[k][d][c] = ew[d][c][k]   (per-tap (N=D, K=H) row-major)
__global__ void pack_embed_w(const float* __restrict__ src, bf16* __restrict__ dst) {
  int i = blockIdx.x * blockDim.x + threadIdx.x;
  const int total = 7 * DD * HH;
  if (i >= total) return;
  int k   = i / (DD * HH);
  int rem = i - k * (DD * HH);
  int d   = rem >> 10;
  int c   = rem & (HH - 1);
  dst[i] = (bf16)src[((size_t)d * HH + c) * 7 + k];
}

__global__ void cast_bf16_kernel(const float* __restrict__ src, bf16* __restrict__ dst, int n) {
  int i = blockIdx.x * blockDim.x + threadIdx.x;
  if (i < n) dst[i] = (bf16)src[i];
}

__global__ void norm_cb(const float* __restrict__ cb, float* __restrict__ cbn) {
  int e = blockIdx.x * blockDim.x + threadIdx.x;
  if (e >= CBN) return;
  const float* r = cb + (size_t)e * CDM;
  float s = 0.f;
#pragma unroll
  for (int c = 0; c < CDM; ++c) s += r[c] * r[c];
  float inv = 1.0f / fmaxf(sqrtf(s), 1e-12f);
  float* o = cbn + (size_t)e * CDM;
#pragma unroll
  for (int c = 0; c < CDM; ++c) o[c] = r[c] * inv;
}

// ---------------- WMMA GEMM: C = A(MxK) @ W(NxK)^T + epilogue ----------------
// block = 256 threads (8 waves); block tile = 256 (M) x 64 (N); wave tile 32x64.
// Per K-step: 2 A-frags + 4 B-frags -> 8 wmma (B frag reused across 2 M-subtiles).
// MODE 0: outf = acc + bias[n]
// MODE 1: outb = bf16(gelu(acc + bias[n]))
// MODE 2: outf += gscale[n] * (acc + bias[n])      (residual, in place)
template <int MODE>
__global__ void gemm_wmma(const bf16* __restrict__ A, const bf16* __restrict__ W,
                          const float* __restrict__ bias, const float* __restrict__ gscale,
                          float* __restrict__ outf, bf16* __restrict__ outb,
                          int M, int N, int K) {
  const int lane = threadIdx.x & 31;
  const int wave = threadIdx.x >> 5;
  const int m0   = blockIdx.y * 256 + wave * 32;
  const int nblk = blockIdx.x * 64;
  const int rowA = m0 + (lane & 15);
  const int ncol = nblk + (lane & 15);

  v8f acc[2][4] = {};
  for (int k0 = 0; k0 < K; k0 += 32) {
    v16bf a0 = load_a_frag(A, K, rowA,      k0, lane);
    v16bf a1 = load_a_frag(A, K, rowA + 16, k0, lane);
#pragma unroll
    for (int s = 0; s < 4; ++s) {
      v16bf b = load_b_frag(W, K, ncol + s * 16, k0, lane);
      acc[0][s] = __builtin_amdgcn_wmma_f32_16x16x32_bf16(false, a0, false, b,
                                                          (short)0, acc[0][s], false, false);
      acc[1][s] = __builtin_amdgcn_wmma_f32_16x16x32_bf16(false, a1, false, b,
                                                          (short)0, acc[1][s], false, false);
    }
  }

#pragma unroll
  for (int s = 0; s < 4; ++s) {
    int col  = ncol + s * 16;
    float bv = bias[col];
#pragma unroll
    for (int i = 0; i < 2; ++i) {
#pragma unroll
      for (int j = 0; j < 8; ++j) {
        int row = m0 + i * 16 + j + ((lane & 16) ? 8 : 0);
        float v = acc[i][s][j] + bv;
        if (MODE == 1) {
          outb[(size_t)row * N + col] = (bf16)gelu_exact(v);
        } else if (MODE == 2) {
          outf[(size_t)row * N + col] += gscale[col] * v;
        } else {
          outf[(size_t)row * N + col] = v;
        }
      }
    }
  }
}

// ---------------- embed conv as 7-tap WMMA GEMM ----------------
// out(MT x D) = sum_k xpad[b, t+k, :](1024) @ wek[k](D x 1024)^T + eb
// wave tile 32x64 (two 16-row A sub-tiles share each B fragment).
__global__ void conv_embed_wmma(const bf16* __restrict__ xpad, const bf16* __restrict__ wek,
                                const float* __restrict__ bias, float* __restrict__ outf) {
  const int lane = threadIdx.x & 31;
  const int wave = threadIdx.x >> 5;
  const int m0   = blockIdx.y * 256 + wave * 32;
  const int nblk = blockIdx.x * 64;
  const int r0   = m0 + (lane & 15);       // first A row for this lane
  const int b0   = r0 >> 11;               // / TT  (32-row tile stays in one batch)
  const int t0   = r0 & (TT - 1);
  const int ncol = nblk + (lane & 15);

  v8f acc[2][4] = {};
  for (int k = 0; k < 7; ++k) {
    const bf16* Arow0 = xpad + ((size_t)(b0 * TP + t0 + k)) * HH;
    const bf16* Arow1 = Arow0 + (size_t)16 * HH;   // row r0+16, same batch
    const bf16* Wk    = wek + (size_t)k * DD * HH;
    for (int k0 = 0; k0 < HH; k0 += 32) {
      const int aoff = k0 + ((lane & 16) ? 8 : 0);
      v16bf a0 = frag_cat(*(const v8bf*)(Arow0 + aoff), *(const v8bf*)(Arow0 + aoff + 16));
      v16bf a1 = frag_cat(*(const v8bf*)(Arow1 + aoff), *(const v8bf*)(Arow1 + aoff + 16));
#pragma unroll
      for (int s = 0; s < 4; ++s) {
        v16bf bb = load_b_frag(Wk, HH, ncol + s * 16, k0, lane);
        acc[0][s] = __builtin_amdgcn_wmma_f32_16x16x32_bf16(false, a0, false, bb,
                                                            (short)0, acc[0][s], false, false);
        acc[1][s] = __builtin_amdgcn_wmma_f32_16x16x32_bf16(false, a1, false, bb,
                                                            (short)0, acc[1][s], false, false);
      }
    }
  }
#pragma unroll
  for (int s = 0; s < 4; ++s) {
    int col  = ncol + s * 16;
    float bv = bias[col];
#pragma unroll
    for (int i = 0; i < 2; ++i) {
#pragma unroll
      for (int j = 0; j < 8; ++j) {
        int row = m0 + i * 16 + j + ((lane & 16) ? 8 : 0);
        outf[(size_t)row * DD + col] = acc[i][s][j] + bv;
      }
    }
  }
}

// ---------------- LayerNorm over D=384 (biased var, eps 1e-6) ----------------
// grid = MT, block = 384.  outf/outb optional.
__global__ void ln_kernel(const float* __restrict__ src, const float* __restrict__ g,
                          const float* __restrict__ be, float* __restrict__ outf,
                          bf16* __restrict__ outb) {
  __shared__ float red[DD];
  const int n = blockIdx.x;
  const int d = threadIdx.x;
  float v = src[(size_t)n * DD + d];

  red[d] = v; __syncthreads();
  if (d < 128) red[d] += red[d + 128] + red[d + 256];
  __syncthreads();
  for (int s = 64; s > 0; s >>= 1) { if (d < s) red[d] += red[d + s]; __syncthreads(); }
  float mean = red[0] * (1.0f / DD);
  __syncthreads();

  red[d] = v * v; __syncthreads();
  if (d < 128) red[d] += red[d + 128] + red[d + 256];
  __syncthreads();
  for (int s = 64; s > 0; s >>= 1) { if (d < s) red[d] += red[d + s]; __syncthreads(); }
  float var  = red[0] * (1.0f / DD) - mean * mean;
  float rstd = rsqrtf(var + 1e-6f);

  float o = (v - mean) * rstd * g[d] + be[d];
  if (outf) outf[(size_t)n * DD + d] = o;
  if (outb) outb[(size_t)n * DD + d] = (bf16)o;
}

// ---------------- fused depthwise conv (k=7, same pad) + LayerNorm -> bf16 ----
__global__ void dwln_kernel(const float* __restrict__ h, const float* __restrict__ dw,
                            const float* __restrict__ db, const float* __restrict__ g,
                            const float* __restrict__ be, bf16* __restrict__ outb) {
  __shared__ float red[DD];
  const int n = blockIdx.x;
  const int t = n & (TT - 1);
  const int d = threadIdx.x;

  float acc = db[d];
#pragma unroll
  for (int k = 0; k < 7; ++k) {
    int tk = t + k - 3;
    if (tk >= 0 && tk < TT)
      acc += h[(size_t)(n + k - 3) * DD + d] * dw[d * 7 + k];
  }

  red[d] = acc; __syncthreads();
  if (d < 128) red[d] += red[d + 128] + red[d + 256];
  __syncthreads();
  for (int s = 64; s > 0; s >>= 1) { if (d < s) red[d] += red[d + s]; __syncthreads(); }
  float mean = red[0] * (1.0f / DD);
  __syncthreads();

  red[d] = acc * acc; __syncthreads();
  if (d < 128) red[d] += red[d + 128] + red[d + 256];
  __syncthreads();
  for (int s = 64; s > 0; s >>= 1) { if (d < s) red[d] += red[d + s]; __syncthreads(); }
  float var  = red[0] * (1.0f / DD) - mean * mean;
  float rstd = rsqrtf(var + 1e-6f);

  outb[(size_t)n * DD + d] = (bf16)((acc - mean) * rstd * g[d] + be[d]);
}

// ---------------- fused VQ: z_e, cosine argmax, dequant projection -----------
// grid = MT, block = 256 (8 waves).
__global__ void vq_kernel(const float* __restrict__ enc, const float* __restrict__ in_w,
                          const float* __restrict__ in_b, const float* __restrict__ cb,
                          const float* __restrict__ cbn, const float* __restrict__ out_w,
                          const float* __restrict__ out_b, float* __restrict__ idx_out,
                          float* __restrict__ q_out) {
  __shared__ float zn[CDM];
  __shared__ float cq[CDM];
  __shared__ float bv[256];
  __shared__ int   bi[256];

  const int n    = blockIdx.x;
  const int tid  = threadIdx.x;
  const int lane = tid & 31;
  const int wave = tid >> 5;
  const float* er = enc + (size_t)n * HH;

  // z_e[wave] = dot(enc row, in_w[wave]) + in_b[wave]
  float a = 0.f;
  for (int j = lane; j < HH; j += 32) a += er[j] * in_w[wave * HH + j];
#pragma unroll
  for (int s = 16; s > 0; s >>= 1) a += __shfl_down(a, s, 32);
  if (lane == 0) zn[wave] = a + in_b[wave];
  __syncthreads();

  if (tid == 0) {
    float s = 0.f;
#pragma unroll
    for (int c = 0; c < CDM; ++c) s += zn[c] * zn[c];
    float inv = 1.0f / fmaxf(sqrtf(s), 1e-12f);
#pragma unroll
    for (int c = 0; c < CDM; ++c) zn[c] *= inv;
  }
  __syncthreads();

  float zr[CDM];
#pragma unroll
  for (int c = 0; c < CDM; ++c) zr[c] = zn[c];

  float best = -3.0e38f; int besti = 0;
  for (int e = tid; e < CBN; e += 256) {
    const float* cr = cbn + (size_t)e * CDM;
    float sim = 0.f;
#pragma unroll
    for (int c = 0; c < CDM; ++c) sim += cr[c] * zr[c];
    if (sim > best) { best = sim; besti = e; }   // strict >: keeps first max
  }
  bv[tid] = best; bi[tid] = besti; __syncthreads();
  for (int s = 128; s > 0; s >>= 1) {
    if (tid < s) {
      if (bv[tid + s] > bv[tid] ||
          (bv[tid + s] == bv[tid] && bi[tid + s] < bi[tid])) {
        bv[tid] = bv[tid + s]; bi[tid] = bi[tid + s];
      }
    }
    __syncthreads();
  }

  if (tid == 0) idx_out[n] = (float)bi[0];
  if (tid < CDM) cq[tid] = cb[(size_t)bi[0] * CDM + tid];
  __syncthreads();

  for (int hh = tid; hh < HH; hh += 256) {
    const float* w = out_w + (size_t)hh * CDM;
    float q = out_b[hh];
#pragma unroll
    for (int c = 0; c < CDM; ++c) q += cq[c] * w[c];
    q_out[(size_t)n * HH + hh] = q;
  }
}

// ---------------- launch ----------------
extern "C" void kernel_launch(void* const* d_in, const int* in_sizes, int n_in,
                              void* d_out, int out_size, void* d_ws, size_t ws_size,
                              hipStream_t stream) {
  const float* x     = (const float*)d_in[0];
  const float* ew    = (const float*)d_in[1];
  const float* eb    = (const float*)d_in[2];
  const float* ng    = (const float*)d_in[3];
  const float* nb    = (const float*)d_in[4];
  const float* dww   = (const float*)d_in[5];
  const float* dwb   = (const float*)d_in[6];
  const float* lng   = (const float*)d_in[7];
  const float* lnb   = (const float*)d_in[8];
  const float* w1    = (const float*)d_in[9];
  const float* b1    = (const float*)d_in[10];
  const float* w2    = (const float*)d_in[11];
  const float* b2    = (const float*)d_in[12];
  const float* gam   = (const float*)d_in[13];
  const float* fg    = (const float*)d_in[14];
  const float* fb    = (const float*)d_in[15];
  const float* ow    = (const float*)d_in[16];
  const float* ob    = (const float*)d_in[17];
  const float* vin_w = (const float*)d_in[18];
  const float* vin_b = (const float*)d_in[19];
  const float* cb    = (const float*)d_in[20];
  const float* vout_w= (const float*)d_in[21];
  const float* vout_b= (const float*)d_in[22];

  char* p = (char*)d_ws;
  auto alloc = [&](size_t bytes) -> void* {
    void* r = (void*)p; p += (bytes + 255) & ~(size_t)255; return r;
  };
  float* h    = (float*)alloc((size_t)MT * DD * 4);
  bf16*  xpad = (bf16*) alloc((size_t)BB * TP * HH * 2);
  bf16*  wek  = (bf16*) alloc((size_t)7 * DD * HH * 2);
  bf16*  w1b  = (bf16*) alloc((size_t)LL * II * DD * 2);
  bf16*  w2b  = (bf16*) alloc((size_t)LL * DD * II * 2);
  bf16*  owb  = (bf16*) alloc((size_t)HH * DD * 2);
  bf16*  ybf  = (bf16*) alloc((size_t)MT * DD * 2);
  bf16*  y2bf = (bf16*) alloc((size_t)MT * II * 2);
  bf16*  hfin = (bf16*) alloc((size_t)MT * DD * 2);
  float* encb = (float*)alloc((size_t)MT * HH * 4);
  float* cbn  = (float*)alloc((size_t)CBN * CDM * 4);

  auto nb256 = [](long n) { return (int)((n + 255) / 256); };

  build_xpad<<<nb256((long)BB * TP * HH), 256, 0, stream>>>(x, xpad);
  pack_embed_w<<<nb256((long)7 * DD * HH), 256, 0, stream>>>(ew, wek);
  cast_bf16_kernel<<<nb256((long)LL * II * DD), 256, 0, stream>>>(w1, w1b, LL * II * DD);
  cast_bf16_kernel<<<nb256((long)LL * DD * II), 256, 0, stream>>>(w2, w2b, LL * DD * II);
  cast_bf16_kernel<<<nb256((long)HH * DD), 256, 0, stream>>>(ow, owb, HH * DD);
  norm_cb<<<nb256(CBN), 256, 0, stream>>>(cb, cbn);

  // embed conv + embed LN
  conv_embed_wmma<<<dim3(DD / 64, MT / 256), 256, 0, stream>>>(xpad, wek, eb, h);
  ln_kernel<<<MT, DD, 0, stream>>>(h, ng, nb, h, nullptr);

  // 12 ConvNeXt blocks
  for (int l = 0; l < LL; ++l) {
    dwln_kernel<<<MT, DD, 0, stream>>>(h, dww + (size_t)l * DD * 7, dwb + (size_t)l * DD,
                                       lng + (size_t)l * DD, lnb + (size_t)l * DD, ybf);
    gemm_wmma<1><<<dim3(II / 64, MT / 256), 256, 0, stream>>>(
        ybf, w1b + (size_t)l * II * DD, b1 + (size_t)l * II, nullptr,
        nullptr, y2bf, MT, II, DD);
    gemm_wmma<2><<<dim3(DD / 64, MT / 256), 256, 0, stream>>>(
        y2bf, w2b + (size_t)l * DD * II, b2 + (size_t)l * DD, gam + (size_t)l * DD,
        h, nullptr, MT, DD, II);
  }

  // final LN + output projection
  ln_kernel<<<MT, DD, 0, stream>>>(h, fg, fb, nullptr, hfin);
  gemm_wmma<0><<<dim3(HH / 64, MT / 256), 256, 0, stream>>>(
      hfin, owb, ob, nullptr, encb, nullptr, MT, HH, DD);

  // VQ: outputs = [idx (MT floats)] ++ [quantized (MT*HH floats)]
  float* idx_out = (float*)d_out;
  float* q_out   = (float*)d_out + MT;
  vq_kernel<<<MT, 256, 0, stream>>>(encb, vin_w, vin_b, cb, cbn, vout_w, vout_b,
                                    idx_out, q_out);
}
